// OperatorModuleT_85650237817645
// MI455X (gfx1250) — matrix operand order repaired and checked by reading, not compiled, not run on hardware
//
#include <hip/hip_runtime.h>

typedef _Float16 v16h __attribute__((ext_vector_type(16)));
typedef _Float16 v8h  __attribute__((ext_vector_type(8)));
typedef _Float16 v4h  __attribute__((ext_vector_type(4)));
typedef _Float16 h2   __attribute__((ext_vector_type(2)));
typedef float    v8f  __attribute__((ext_vector_type(8)));
typedef unsigned int u32x4 __attribute__((ext_vector_type(4)));
typedef int          i32x4 __attribute__((ext_vector_type(4)));
typedef int          i32x8 __attribute__((ext_vector_type(8)));

#define N_ANGLES 720
#define DET      512
#define GRID_N   512
#define BATCH    16
#define PAD      112            // halves of zero-pad before det index 0
#define STRIDEH  760            // halves per batch row in LDS (mult of 8): covers d in [-112, 648)
#define LDS_HALVES (BATCH * STRIDEH)   // 12160 halves per pane
#define TILES    8

#if __has_builtin(__builtin_amdgcn_tensor_load_to_lds)
#define HAVE_TDM 1
#else
#define HAVE_TDM 0
#endif

#if HAVE_TDM
// One TDM descriptor: 2D tile 512(det) x 16(batch) f16, global row stride = 720*512 elements,
// LDS rows padded 512 halves (256 DWORDs) + 124 DWORDs -> 760-half row stride.
__device__ __forceinline__ void tdm_issue_slice(unsigned lds_byte_addr, const _Float16* gptr) {
    unsigned long long ga = (unsigned long long)(size_t)gptr;
    u32x4 g0;
    g0[0] = 1u;                                   // count=1 (valid user descriptor)
    g0[1] = lds_byte_addr;                        // LDS dest (bytes), includes +PAD*2
    g0[2] = (unsigned)ga;                         // global_addr[31:0]
    g0[3] = (unsigned)((ga >> 32) & 0x01FFFFFFull) | 0x80000000u;  // addr[56:32] | type=2
    i32x8 g1;
    g1[0] = (int)((1u << 16)      // data_size = 1 (2 bytes)
                | (1u << 20)      // pad_enable
                | (7u << 22)      // pad_interval: every 256 DWORDs (= one 512-half row)
                | (123u << 25));  // pad_amount: 124 DWORDs (row stride -> 760 halves)
    g1[1] = (int)(512u << 16);    // tensor_dim0[15:0] = 512 (bits 63:48)
    g1[2] = (int)(16u  << 16);    // tensor_dim1[15:0] = 16  (bits 95:80)
    g1[3] = (int)(512u << 16);    // tile_dim0 = 512 (bits 127:112)
    g1[4] = (int)(16u);           // tile_dim1 = 16, tile_dim2 = 0
    g1[5] = (int)(368640u);       // tensor_dim0_stride = 720*512 elements (batch stride)
    g1[6] = 0;
    g1[7] = 0;
    i32x4 z4 = {0, 0, 0, 0};
#if defined(__clang_major__) && __clang_major__ >= 23
    i32x8 z8 = {0, 0, 0, 0, 0, 0, 0, 0};
    __builtin_amdgcn_tensor_load_to_lds(g0, g1, z4, z4, z8, 0);
#else
    __builtin_amdgcn_tensor_load_to_lds(g0, g1, z4, z4, 0);
#endif
}

__device__ __forceinline__ void tdm_wait0() {
#if __has_builtin(__builtin_amdgcn_s_wait_tensorcnt)
    __builtin_amdgcn_s_wait_tensorcnt(0);
#else
    asm volatile("s_wait_tensorcnt 0x0" ::: "memory");
#endif
}
#endif

// One-time f32 -> f16 sinogram conversion into workspace (same [B][A][D] layout).
__global__ __launch_bounds__(256)
void cvt_f16(const float* __restrict__ in, _Float16* __restrict__ out) {
    const size_t i = ((size_t)blockIdx.x * 256 + threadIdx.x) * 4;
    const float4 v = *(const float4*)(in + i);
    v4h h;
    h[0] = (_Float16)v.x; h[1] = (_Float16)v.y;
    h[2] = (_Float16)v.z; h[3] = (_Float16)v.w;
    *(v4h*)(out + i) = h;
}

template <bool USE_TDM>
__global__ __launch_bounds__(256)
void bp_wmma_f16(const float* __restrict__ sino32,
                 const _Float16* __restrict__ sino16,
                 const float* __restrict__ angles,
                 float* __restrict__ out)
{
    __shared__ _Float16 sbuf[2][LDS_HALVES];   // 48,640 B, zero-padded borders
    __shared__ float2   cstab[N_ANGLES];       //  5,760 B

    const int t = threadIdx.x;

    // Zero both panes once: pads stay zero => OOB taps contribute exactly 0.
    {
        unsigned int* p = (unsigned int*)&sbuf[0][0];
        for (int i = t; i < LDS_HALVES; i += 256) p[i] = 0u;   // 2 panes = 12160 dwords
    }
    for (int i = t; i < N_ANGLES; i += 256) {
        float th = angles[i];
        cstab[i] = make_float2(cosf(th), sinf(th));
    }

    const int lane   = t & 31;
    const int wv     = t >> 5;          // wave id 0..7 -> image row within block
    const int n      = lane & 15;       // batch column (B,C) == pixel row m (A)
    const int hiHalf = lane >> 4;
    const int kbaseA = hiHalf * 8;      // A lane K-group
    const int koffB  = hiHalf * 8;      // B lane K-group
    const int moff   = hiHalf * 8;      // C/D row offset

    const int x0 = blockIdx.x * 128;
    const int y  = blockIdx.y * 8 + wv;

    // Fallback staging assignment
    const int sb = t >> 4;
    const int sl = t & 15;

    v8f acc[TILES];
    #pragma unroll
    for (int i = 0; i < TILES; ++i) acc[i] = v8f{};

    float4 rg[8];
    if constexpr (!USE_TDM) {
        const float4* row = (const float4*)(sino32 + ((size_t)sb * N_ANGLES + 0) * DET);
        #pragma unroll
        for (int j = 0; j < 8; ++j) rg[j] = row[sl + j * 16];
    }
    __syncthreads();   // zero-init + cos/sin table ready

#if HAVE_TDM
    if constexpr (USE_TDM) {
        if (wv == 0) {
            const unsigned lds0 = (unsigned)(size_t)(&sbuf[0][0]) + PAD * 2;
            tdm_issue_slice(lds0, sino16 + (size_t)0 * DET);
        }
    }
#endif

    const float xc = (float)x0 - 255.5f;
    const float yc = 255.5f - (float)y;
    const float mF = (float)n;
    const h2 h_one  = {(_Float16)1.0f, (_Float16)1.0f};
    const h2 h_zero = {(_Float16)0.0f, (_Float16)0.0f};

    for (int a = 0; a < N_ANGLES; ++a) {
        const int buf = a & 1;

        if constexpr (!USE_TDM) {
            // Store staged regs f32 -> f16 into current pane
            _Float16* dst = &sbuf[buf][sb * STRIDEH + PAD];
            #pragma unroll
            for (int j = 0; j < 8; ++j) {
                const int d = (sl + j * 16) * 4;
                v4h h;
                h[0] = (_Float16)rg[j].x; h[1] = (_Float16)rg[j].y;
                h[2] = (_Float16)rg[j].z; h[3] = (_Float16)rg[j].w;
                *(v4h*)(dst + d) = h;
            }
            __syncthreads();
            if (a + 1 < N_ANGLES) {   // register-prefetch next angle under compute
                const float4* row = (const float4*)(sino32 + ((size_t)sb * N_ANGLES + (a + 1)) * DET);
                #pragma unroll
                for (int j = 0; j < 8; ++j) rg[j] = row[sl + j * 16];
            }
        }
#if HAVE_TDM
        if constexpr (USE_TDM) {
            if (wv == 0) tdm_wait0();     // my DMA for pane `buf` done
            __syncthreads();              // completion visible to all waves; pane^1 free
            if (wv == 0 && a + 1 < N_ANGLES) {
                const unsigned ldsn = (unsigned)(size_t)(&sbuf[buf ^ 1][0]) + PAD * 2;
                tdm_issue_slice(ldsn, sino16 + (size_t)(a + 1) * DET);
            }
        }
#endif

        const float2 cs = cstab[a];
        const float c = cs.x, s = cs.y;
        const float u00  = xc * c + yc * s + 255.5f;
        const float mc   = mF * c;
        const float minc = fminf(0.0f, 15.0f * c);
        const float step = 16.0f * c;
        float ut = u00;

        #pragma unroll
        for (int tl = 0; tl < TILES; ++tl) {
            const int dbase = ((int)floorf(ut + minc)) & ~7;       // window base, 8-aligned
            const float up = ut + mc - (float)dbase - (float)kbaseA;

            // A[m][k] = max(0, 1 - |u_m - (dbase+k)|): tent = 2-tap linear interp weights.
            // t in f32 (needs range), weight math in packed f16 (exact where |t|<1, clamps elsewhere).
            v16h A;
            #pragma unroll
            for (int ep = 0; ep < 8; ++ep) {
                const int k0 = (ep < 4) ? (2 * ep) : (2 * ep + 8);
                const float ta = up - (float)k0;
                const float tb = up - (float)(k0 + 1);
                h2 th; th[0] = (_Float16)ta; th[1] = (_Float16)tb;
                h2 w = h_one - __builtin_elementwise_abs(th);
                w = __builtin_elementwise_max(w, h_zero);
                A[2 * ep]     = w[0];
                A[2 * ep + 1] = w[1];
            }

            // B[k][n] from LDS [batch][det]: 2x 16B-aligned ds_load_b128 per lane
            const _Float16* bp = &sbuf[buf][n * STRIDEH + PAD + dbase + koffB];
            const v8h blo = *(const v8h*)bp;
            const v8h bhi = *(const v8h*)(bp + 16);
            v16h B;
            #pragma unroll
            for (int e = 0; e < 8; ++e) { B[e] = blo[e]; B[e + 8] = bhi[e]; }

            acc[tl] = __builtin_amdgcn_wmma_f32_16x16x32_f16(
                false, A, false, B, (short)0, acc[tl], false, false);

            ut += step;
        }
    }

    // C/D layout: element r of acc = pixel M = moff + r, lane's batch = n
    #pragma unroll
    for (int tl = 0; tl < TILES; ++tl) {
        float* op = out + ((size_t)n * GRID_N + y) * GRID_N + (x0 + tl * 16 + moff);
        float4 lo, hi;
        lo.x = acc[tl][0]; lo.y = acc[tl][1]; lo.z = acc[tl][2]; lo.w = acc[tl][3];
        hi.x = acc[tl][4]; hi.y = acc[tl][5]; hi.z = acc[tl][6]; hi.w = acc[tl][7];
        *(float4*)op       = lo;
        *(float4*)(op + 4) = hi;
    }
}

extern "C" void kernel_launch(void* const* d_in, const int* in_sizes, int n_in,
                              void* d_out, int out_size, void* d_ws, size_t ws_size,
                              hipStream_t stream) {
    const float* x      = (const float*)d_in[0];
    const float* angles = (const float*)d_in[1];
    float* out          = (float*)d_out;
    (void)in_sizes; (void)n_in; (void)out_size;

    dim3 grid(GRID_N / 128, GRID_N / 8, 1);   // 4 x 64 workgroups
    dim3 block(256, 1, 1);                    // 8 wave32 waves

    const size_t n_sino = (size_t)BATCH * N_ANGLES * DET;
#if HAVE_TDM
    if (ws_size >= n_sino * sizeof(_Float16)) {
        _Float16* s16 = (_Float16*)d_ws;
        cvt_f16<<<dim3((unsigned)(n_sino / (256 * 4))), block, 0, stream>>>(x, s16);
        bp_wmma_f16<true><<<grid, block, 0, stream>>>(x, s16, angles, out);
        return;
    }
#endif
    bp_wmma_f16<false><<<grid, block, 0, stream>>>(x, nullptr, angles, out);
}
